// HybridAttentionLayer_44083544326405
// MI455X (gfx1250) — matrix operand order repaired and checked by reading, not compiled, not run on hardware
//
#include <hip/hip_runtime.h>
#include <cstdint>
#include <cstddef>

#define DEV __device__ __forceinline__

typedef __attribute__((ext_vector_type(8)))  float  v8f;
typedef __attribute__((ext_vector_type(16))) __bf16 bf16x16;
typedef __attribute__((ext_vector_type(8)))  __bf16 bf16x8;

static constexpr int T_  = 4096;
static constexpr int D_  = 1024;
static constexpr int H_  = 16;
static constexpr int HD_ = 64;
static constexpr int WLOC_ = 256;

DEV bf16x8 ld8(const __bf16* p){ return *(const bf16x8*)p; }
DEV bf16x16 cat16(bf16x8 a, bf16x8 b){
  return __builtin_shufflevector(a, b, 0,1,2,3,4,5,6,7,8,9,10,11,12,13,14,15);
}
DEV v8f zero8(){ v8f z = {0.f,0.f,0.f,0.f,0.f,0.f,0.f,0.f}; return z; }
DEV v8f wmma_bf16(bf16x16 a, bf16x16 b, v8f c){
  // D = A(16x32 bf16) x B(32x16 bf16) + C(16x16 f32)
  return __builtin_amdgcn_wmma_f32_16x16x32_bf16(false, a, false, b, (short)0, c, false, false);
}
DEV void store_out(float* p, float v){ *p = v; }
DEV void store_out(__bf16* p, float v){ *p = (__bf16)v; }

// ---------------------------------------------------------------------------
// 1) fp32 -> bf16 convert (vectorized x4)
// ---------------------------------------------------------------------------
__global__ void cvt_f32_to_bf16(const float* __restrict__ in, __bf16* __restrict__ out, int n){
  int i = (blockIdx.x * blockDim.x + threadIdx.x) * 4;
  if (i < n) {
    float4 v = *(const float4*)(in + i);
    out[i+0] = (__bf16)v.x; out[i+1] = (__bf16)v.y;
    out[i+2] = (__bf16)v.z; out[i+3] = (__bf16)v.w;
  }
}

// ---------------------------------------------------------------------------
// 2) W (K x N, fp32, row-major) -> W^T (N x K, bf16, K contiguous)
// ---------------------------------------------------------------------------
__global__ void transpose_to_bf16(const float* __restrict__ in, __bf16* __restrict__ out,
                                  int K, int N){
  __shared__ float tile[32][33];
  int nb = blockIdx.x * 32, kb = blockIdx.y * 32;
  int tx = threadIdx.x & 31, ty = threadIdx.x >> 5;   // 256 threads: ty 0..7
  #pragma unroll
  for (int i = 0; i < 32; i += 8)
    tile[ty + i][tx] = in[(size_t)(kb + ty + i) * N + (nb + tx)];
  __syncthreads();
  #pragma unroll
  for (int i = 0; i < 32; i += 8)
    out[(size_t)(nb + ty + i) * K + (kb + tx)] = (__bf16)tile[tx][ty + i];
}

// ---------------------------------------------------------------------------
// 3) Gate GEMV: g[t] = sigmoid(hidden[t,:] . Wg + bg)   (fp32, one wave/token)
// ---------------------------------------------------------------------------
__global__ void gate_kernel(const float* __restrict__ hs, const float* __restrict__ Wg,
                            const float* __restrict__ bg, float* __restrict__ g){
  int wv = blockIdx.x * 8 + (threadIdx.x >> 5);
  int lane = threadIdx.x & 31;
  const float* row = hs + (size_t)wv * D_;
  float s = 0.f;
  #pragma unroll 8
  for (int i = lane; i < D_; i += 32) s += row[i] * Wg[i];
  #pragma unroll
  for (int o = 16; o > 0; o >>= 1) s += __shfl_xor(s, o, 32);
  if (lane == 0) g[wv] = 1.f / (1.f + __expf(-(s + bg[0])));
}

// ---------------------------------------------------------------------------
// 4) WMMA GEMM: C(MxN) = A(MxK,bf16) @ Bt(NxK,bf16)^T + bias
//    128x128 block, BK=32, 4 waves (2x2), 64x64 per wave -> 16 WMMA/K-step,
//    double-buffered LDS (one barrier per K-step) + global_prefetch 2 steps ahead
// ---------------------------------------------------------------------------
template <typename OutT>
__global__ __launch_bounds__(128) void gemm_bf16(const __bf16* __restrict__ A,
                                                 const __bf16* __restrict__ Bt,
                                                 const float* __restrict__ bias,
                                                 OutT* __restrict__ C,
                                                 int M, int N, int K){
  constexpr int BK = 32;
  __shared__ __attribute__((aligned(16))) __bf16 As[2][128][BK];
  __shared__ __attribute__((aligned(16))) __bf16 Bs[2][128][BK];
  int tid = threadIdx.x;
  int wave = tid >> 5, lane = tid & 31;
  int half = lane >> 4, ml = lane & 15;
  int wm = wave >> 1, wn = wave & 1;            // 2x2 waves, 64x64 per wave
  int m0 = blockIdx.y * 128, n0 = blockIdx.x * 128;

  v8f acc[4][4];
  #pragma unroll
  for (int i = 0; i < 4; i++)
    #pragma unroll
    for (int j = 0; j < 4; j++) acc[i][j] = zero8();

  // per-thread staging: 4 x 16B chunks of A + 4 of B  (chunk c = tid + i*128)
  bf16x8 ra[4], rb[4];
  #pragma unroll
  for (int i = 0; i < 4; i++) {
    int c = tid + i * 128, row = c >> 2, ch = c & 3;
    ra[i] = ld8(A  + (size_t)(m0 + row) * K + ch * 8);
    rb[i] = ld8(Bt + (size_t)(n0 + row) * K + ch * 8);
  }
  #pragma unroll
  for (int i = 0; i < 4; i++) {
    int c = tid + i * 128, row = c >> 2, ch = c & 3;
    *(bf16x8*)&As[0][row][ch * 8] = ra[i];
    *(bf16x8*)&Bs[0][row][ch * 8] = rb[i];
  }
  __syncthreads();

  int buf = 0;
  for (int k0 = 0; k0 < K; k0 += BK) {
    bool more = (k0 + BK) < K;
    if (more) {                                 // issue next tile's global loads
      #pragma unroll
      for (int i = 0; i < 4; i++) {
        int c = tid + i * 128, row = c >> 2, ch = c & 3;
        ra[i] = ld8(A  + (size_t)(m0 + row) * K + k0 + BK + ch * 8);
        rb[i] = ld8(Bt + (size_t)(n0 + row) * K + k0 + BK + ch * 8);
      }
    }
    if (k0 + 2 * BK < K) {                      // prefetch tile after next
      __builtin_prefetch(A  + (size_t)(m0 + tid) * K + k0 + 2 * BK, 0, 1);
      __builtin_prefetch(Bt + (size_t)(n0 + tid) * K + k0 + 2 * BK, 0, 1);
    }

    bf16x16 af[4], bfr[4];
    #pragma unroll
    for (int i = 0; i < 4; i++) {               // A frag: lane=row, K split
      int row = wm * 64 + i * 16 + ml;
      af[i] = cat16(ld8(&As[buf][row][half * 8]), ld8(&As[buf][row][16 + half * 8]));
    }
    #pragma unroll
    for (int j = 0; j < 4; j++) {               // B frag: lane=col, K split
      int row = wn * 64 + j * 16 + ml;
      bfr[j] = cat16(ld8(&Bs[buf][row][half * 8]), ld8(&Bs[buf][row][16 + half * 8]));
    }
    #pragma unroll
    for (int i = 0; i < 4; i++)
      #pragma unroll
      for (int j = 0; j < 4; j++)
        acc[i][j] = wmma_bf16(af[i], bfr[j], acc[i][j]);

    if (more) {                                 // stage into the other buffer
      #pragma unroll
      for (int i = 0; i < 4; i++) {
        int c = tid + i * 128, row = c >> 2, ch = c & 3;
        *(bf16x8*)&As[buf ^ 1][row][ch * 8] = ra[i];
        *(bf16x8*)&Bs[buf ^ 1][row][ch * 8] = rb[i];
      }
    }
    __syncthreads();
    buf ^= 1;
  }

  #pragma unroll
  for (int i = 0; i < 4; i++)
    #pragma unroll
    for (int j = 0; j < 4; j++) {
      int col = n0 + wn * 64 + j * 16 + ml;
      float b = bias ? bias[col] : 0.f;
      #pragma unroll
      for (int r = 0; r < 8; r++) {
        int row = m0 + wm * 64 + i * 16 + r + 8 * half;  // C layout: M = r + 8*half
        store_out(&C[(size_t)row * N + col], acc[i][j][r] + b);
      }
    }
}

// ---------------------------------------------------------------------------
// 5) Repack qkv (T x 3D, bf16) -> Q[H][T][HD] (pre-scaled), K[H][T][HD], V^T[H][HD][T]
// ---------------------------------------------------------------------------
__global__ void repack_qkv(const __bf16* __restrict__ qkv, __bf16* __restrict__ Q,
                           __bf16* __restrict__ Kk, __bf16* __restrict__ Vt){
  int idx = blockIdx.x * 256 + threadIdx.x;       // T*D threads
  int t = idx >> 10, c = idx & 1023;
  int h = c >> 6, hd = c & 63;
  const __bf16* row = qkv + (size_t)t * (3 * D_);
  float qv = (float)row[c] * 0.125f;              // fold 1/sqrt(64) into Q
  Q [((size_t)h * T_ + t) * HD_ + hd] = (__bf16)qv;
  Kk[((size_t)h * T_ + t) * HD_ + hd] = row[D_ + c];
  Vt[((size_t)h * HD_ + hd) * T_ + t] = row[2 * D_ + c];
}

// ---------------------------------------------------------------------------
// 6) Flash attention (local band + global-64), 1 wave = 16 queries of one head
// ---------------------------------------------------------------------------
DEV void flash_block(int t0, int q0, bool domask,
                     const bf16x16& aq0, const bf16x16& aq1,
                     const __bf16* __restrict__ Kh, const __bf16* __restrict__ Vth,
                     __bf16* Pw, int lane,
                     v8f* o, float* mrow, float* lrow){
  int half = lane >> 4, ml = lane & 15;
  // QK^T: two 16-key score tiles, K-dim = HD=64 -> 2 chained WMMA each
  const __bf16* k0p = Kh + (size_t)(t0 + ml)      * HD_ + half * 16;
  const __bf16* k1p = Kh + (size_t)(t0 + 16 + ml) * HD_ + half * 16;
  v8f s0 = zero8(), s1 = zero8();
  s0 = wmma_bf16(aq0, cat16(ld8(k0p),      ld8(k0p + 8)),  s0);
  s0 = wmma_bf16(aq1, cat16(ld8(k0p + 32), ld8(k0p + 40)), s0);
  s1 = wmma_bf16(aq0, cat16(ld8(k1p),      ld8(k1p + 8)),  s1);
  s1 = wmma_bf16(aq1, cat16(ld8(k1p + 32), ld8(k1p + 40)), s1);

  if (domask) {
    int t0m = t0 + ml, t1m = t0 + 16 + ml;
    #pragma unroll
    for (int r = 0; r < 8; r++) {
      int q = q0 + r + 8 * half;
      int d0 = q - t0m; if (d0 < -WLOC_ || d0 > WLOC_) s0[r] = -1e30f;
      int d1 = q - t1m; if (d1 < -WLOC_ || d1 > WLOC_) s1[r] = -1e30f;
    }
  }
  float alpha[8];
  #pragma unroll
  for (int r = 0; r < 8; r++) {
    float x = fmaxf(s0[r], s1[r]);                 // row-max within 16-lane half
    x = fmaxf(x, __shfl_xor(x, 1, 32));
    x = fmaxf(x, __shfl_xor(x, 2, 32));
    x = fmaxf(x, __shfl_xor(x, 4, 32));
    x = fmaxf(x, __shfl_xor(x, 8, 32));
    float mn = fmaxf(mrow[r], x);
    alpha[r] = __expf(mrow[r] - mn);
    mrow[r] = mn;
    float p0 = __expf(s0[r] - mn), p1 = __expf(s1[r] - mn);
    s0[r] = p0; s1[r] = p1;
    float sm = p0 + p1;
    sm += __shfl_xor(sm, 1, 32); sm += __shfl_xor(sm, 2, 32);
    sm += __shfl_xor(sm, 4, 32); sm += __shfl_xor(sm, 8, 32);
    lrow[r] = lrow[r] * alpha[r] + sm;
  }
  // C-layout P -> LDS (16x32 bf16), re-read in A-layout (same-wave DS is in-order)
  #pragma unroll
  for (int r = 0; r < 8; r++) {
    int row = r + 8 * half;
    Pw[row * 32 + ml]      = (__bf16)s0[r];
    Pw[row * 32 + 16 + ml] = (__bf16)s1[r];
  }
  __builtin_amdgcn_wave_barrier();
  bf16x16 ap = cat16(ld8(Pw + ml * 32 + half * 16),
                     ld8(Pw + ml * 32 + half * 16 + 8));
  __builtin_amdgcn_wave_barrier();
  #pragma unroll
  for (int j = 0; j < 4; j++) {                    // P(16x32) @ V(32x16) per hd tile
    #pragma unroll
    for (int r = 0; r < 8; r++) o[j][r] *= alpha[r];
    const __bf16* vp = Vth + (size_t)(j * 16 + ml) * T_ + t0 + half * 16;
    o[j] = wmma_bf16(ap, cat16(ld8(vp), ld8(vp + 8)), o[j]);
  }
}

__global__ __launch_bounds__(128) void attn_kernel(const __bf16* __restrict__ Q,
                                                   const __bf16* __restrict__ Km,
                                                   const __bf16* __restrict__ Vt,
                                                   const float* __restrict__ gate,
                                                   __bf16* __restrict__ attn){
  __shared__ __attribute__((aligned(16))) __bf16 Pb[4][16 * 32];
  int wave = threadIdx.x >> 5, lane = threadIdx.x & 31;
  int half = lane >> 4, ml = lane & 15;
  int h = blockIdx.y;
  int q0 = (blockIdx.x * 4 + wave) * 16;
  const __bf16* Kh  = Km + (size_t)h * T_ * HD_;
  const __bf16* Vth = Vt + (size_t)h * HD_ * T_;

  // Q A-fragments (hd 0..31 and 32..63); Q already scaled by 1/sqrt(HD)
  const __bf16* qrow = Q + ((size_t)h * T_ + q0 + ml) * HD_ + half * 16;
  bf16x16 aq0 = cat16(ld8(qrow),      ld8(qrow + 8));
  bf16x16 aq1 = cat16(ld8(qrow + 32), ld8(qrow + 40));

  v8f ol[4], og[4];
  float mlr[8], llr[8], mgr[8], lgr[8];
  #pragma unroll
  for (int j = 0; j < 4; j++) { ol[j] = zero8(); og[j] = zero8(); }
  #pragma unroll
  for (int r = 0; r < 8; r++) { mlr[r] = -1e30f; llr[r] = 0.f; mgr[r] = -1e30f; lgr[r] = 0.f; }

  // local banded attention: keys in [q0-256, q0+15+256]
  int tlo = q0 - WLOC_; if (tlo < 0) tlo = 0; tlo &= ~31;
  int thi = q0 + 16 + WLOC_; if (thi > T_) thi = T_;
  for (int t0 = tlo; t0 < thi; t0 += 32)
    flash_block(t0, q0, true, aq0, aq1, Kh, Vth, Pb[wave], lane, ol, mlr, llr);

  // global attention over first 64 keys (unmasked)
  flash_block(0,  q0, false, aq0, aq1, Kh, Vth, Pb[wave], lane, og, mgr, lgr);
  flash_block(32, q0, false, aq0, aq1, Kh, Vth, Pb[wave], lane, og, mgr, lgr);

  // gate-mix, merge heads back to [T][D]
  #pragma unroll
  for (int r = 0; r < 8; r++) {
    int q = q0 + r + 8 * half;
    float gq = gate[q];
    float il = 1.f / llr[r], ig = 1.f / lgr[r];
    #pragma unroll
    for (int j = 0; j < 4; j++) {
      float val = (1.f - gq) * ol[j][r] * il + gq * og[j][r] * ig;
      attn[(size_t)q * D_ + h * HD_ + j * 16 + ml] = (__bf16)val;
    }
  }
}

// ---------------------------------------------------------------------------
// Launch
// ---------------------------------------------------------------------------
extern "C" void kernel_launch(void* const* d_in, const int* in_sizes, int n_in,
                              void* d_out, int out_size, void* d_ws, size_t ws_size,
                              hipStream_t stream) {
  const float* hs   = (const float*)d_in[0];   // [T,D]
  const float* Wqkv = (const float*)d_in[1];   // [D,3D]
  const float* bqkv = (const float*)d_in[2];   // [3D]
  const float* Wout = (const float*)d_in[3];   // [D,D]
  const float* bout = (const float*)d_in[4];   // [D]
  const float* Wg   = (const float*)d_in[5];   // [D,1]
  const float* bg   = (const float*)d_in[6];   // [1]
  float* out = (float*)d_out;                  // [T,D] fp32

  char* ws = (char*)d_ws;
  __bf16* Xb    = (__bf16*)ws; ws += (size_t)T_ * D_ * 2;
  __bf16* WqkvT = (__bf16*)ws; ws += (size_t)3 * D_ * D_ * 2;
  __bf16* WoutT = (__bf16*)ws; ws += (size_t)D_ * D_ * 2;
  __bf16* qkv   = (__bf16*)ws; ws += (size_t)T_ * 3 * D_ * 2;
  __bf16* Qs    = (__bf16*)ws; ws += (size_t)H_ * T_ * HD_ * 2;
  __bf16* Kb    = (__bf16*)ws; ws += (size_t)H_ * T_ * HD_ * 2;
  __bf16* Vtb   = (__bf16*)ws; ws += (size_t)H_ * HD_ * T_ * 2;
  __bf16* attn  = (__bf16*)ws; ws += (size_t)T_ * D_ * 2;
  float*  gbuf  = (float*)ws;  ws += (size_t)T_ * 4;

  // prep: bf16 convert + weight transposes + gate
  cvt_f32_to_bf16<<<(T_ * D_) / 1024, 256, 0, stream>>>(hs, Xb, T_ * D_);
  transpose_to_bf16<<<dim3(3 * D_ / 32, D_ / 32), 256, 0, stream>>>(Wqkv, WqkvT, D_, 3 * D_);
  transpose_to_bf16<<<dim3(D_ / 32, D_ / 32), 256, 0, stream>>>(Wout, WoutT, D_, D_);
  gate_kernel<<<T_ / 8, 256, 0, stream>>>(hs, Wg, bg, gbuf);

  // QKV GEMM (bf16 out) -> repack to attention layouts
  gemm_bf16<__bf16><<<dim3(3 * D_ / 128, T_ / 128), 128, 0, stream>>>(
      Xb, WqkvT, bqkv, qkv, T_, 3 * D_, D_);
  repack_qkv<<<(T_ * D_) / 256, 256, 0, stream>>>(qkv, Qs, Kb, Vtb);

  // fused local+global flash attention + gate mix
  attn_kernel<<<dim3(T_ / 64, H_), 128, 0, stream>>>(Qs, Kb, Vtb, gbuf, attn);

  // output projection (fp32 out + bias)
  gemm_bf16<float><<<dim3(D_ / 128, T_ / 128), 128, 0, stream>>>(
      attn, WoutT, bout, out, T_, D_, D_);
}
